// Graphsage_Model_21895743275046
// MI455X (gfx1250) — matrix operand order repaired and checked by reading, not compiled, not run on hardware
//
#include <hip/hip_runtime.h>
#include <hip/hip_bf16.h>
#include <math.h>

// ---------------------------------------------------------------------------
// GraphSAGE/GCN retrieval model for MI455X (gfx1250, wave32, WMMA + TDM).
// GEMMs use v_wmma_f32_16x16x32_bf16 (bf16 in, f32 accumulate); the A tile of
// the node-feature GEMM is staged into LDS by the Tensor Data Mover.
// ---------------------------------------------------------------------------

typedef __bf16 bf16;
typedef __attribute__((ext_vector_type(16))) __bf16 v16bf;
typedef __attribute__((ext_vector_type(8)))  __bf16 v8bf;
typedef __attribute__((ext_vector_type(8)))  float  v8f;

typedef __attribute__((ext_vector_type(4))) unsigned int u32x4;
typedef __attribute__((ext_vector_type(8))) int          i32x8;
typedef __attribute__((ext_vector_type(4))) int          i32x4;

#define DDIM 768
#define TOPK 20

__device__ __forceinline__ v16bf cat8(v8bf lo, v8bf hi) {
  v16bf r;
#pragma unroll
  for (int i = 0; i < 8; ++i) { r[i] = lo[i]; r[i + 8] = hi[i]; }
  return r;
}

// ----------------------------- small utility kernels -----------------------

__global__ void cvt_bf16_kernel(const float* __restrict__ in, bf16* __restrict__ out,
                                long n, int relu) {
  long i = (long)blockIdx.x * blockDim.x + threadIdx.x;
  long stride = (long)gridDim.x * blockDim.x;
  for (; i < n; i += stride) {
    float v = in[i];
    if (relu) v = v > 0.f ? v : 0.f;
    out[i] = (bf16)v;
  }
}

__global__ void deg_init_kernel(float* __restrict__ deg, int n) {
  int i = blockIdx.x * blockDim.x + threadIdx.x;
  if (i < n) deg[i] = 1.0f;  // self-loop
}

__global__ void deg_edge_kernel(const int* __restrict__ dst, float* __restrict__ deg, int e) {
  int i = blockIdx.x * blockDim.x + threadIdx.x;
  if (i < e) atomicAdd(&deg[dst[i]], 1.0f);
}

__global__ void dinv_kernel(const float* __restrict__ deg, float* __restrict__ dinv, int n) {
  int i = blockIdx.x * blockDim.x + threadIdx.x;
  if (i < n) dinv[i] = rsqrtf(deg[i]);
}

// G[i,:] = bias + H[i,:] * dinv[i]^2   (self-loop message + bias)
__global__ void ginit_kernel(const float* __restrict__ H, const float* __restrict__ dinv,
                             const float* __restrict__ bias, float* __restrict__ G, long total) {
  long idx = (long)blockIdx.x * blockDim.x + threadIdx.x;
  long stride = (long)gridDim.x * blockDim.x;
  for (; idx < total; idx += stride) {
    long i = idx / DDIM;
    int  j = (int)(idx % DDIM);
    float di = dinv[i];
    G[idx] = bias[j] + H[idx] * di * di;
  }
}

// one wave per edge: G[dst,:] += H[src,:] * dinv[src]*dinv[dst]
__global__ __launch_bounds__(256) void scatter_kernel(const float* __restrict__ H,
                                                      float* __restrict__ G,
                                                      const int* __restrict__ src,
                                                      const int* __restrict__ dst,
                                                      const float* __restrict__ dinv, int E) {
  int wid = threadIdx.x >> 5, lane = threadIdx.x & 31;
  int e = blockIdx.x * 8 + wid;
  if (e >= E) return;
  int s = src[e], d = dst[e];
  float nrm = dinv[s] * dinv[d];
  const float* hr = H + (size_t)s * DDIM;
  float* gr = G + (size_t)d * DDIM;
  __builtin_prefetch(hr, 0, 1);  // global_prefetch_b8
#pragma unroll
  for (int c = 0; c < 6; ++c) {
    int j = c * 128 + lane * 4;
    float4 v = *(const float4*)(hr + j);
    atomicAdd(gr + j + 0, v.x * nrm);
    atomicAdd(gr + j + 1, v.y * nrm);
    atomicAdd(gr + j + 2, v.z * nrm);
    atomicAdd(gr + j + 3, v.w * nrm);
  }
}

// ------------------- TDM: DMA a 64x32 bf16 tile (row stride 768) to LDS ----
// D# per CDNA5 ISA ch.8: group0 = {count/flags, lds_addr, global_addr, type},
// group1 = {mask/data_size, tensor dims, tile dims, strides}. OOB rows beyond
// tensor_dim1 are zero-filled by hardware (handles the ragged last M tile).
// This toolchain exposes the 6-arg builtin:
//   (uint32x4 g0, int32x8 g1, int32x4 g2, int32x4 g3, int32x8 extra, i32 cpol)

#if __has_builtin(__builtin_amdgcn_tensor_load_to_lds)
#define HAVE_TDM 1
__device__ __forceinline__ void tdm_load_tile_64x32_bf16(const bf16* gptr,
                                                         unsigned lds_off, int rows) {
  unsigned long long ga = (unsigned long long)(uintptr_t)gptr;
  u32x4 g0;
  g0[0] = 1u;                                            // count=1 (valid user D#)
  g0[1] = lds_off;                                       // lds_addr (bytes)
  g0[2] = (unsigned)(ga & 0xffffffffu);                  // global_addr[31:0]
  g0[3] = (unsigned)((ga >> 32) & 0x01ffffffu)           // global_addr[56:32]
          | (2u << 30);                                  // type=2 ("image")
  i32x8 g1;
  g1[0] = 0x00010000;                  // workgroup_mask=0, data_size=1 (2 bytes)
  g1[1] = (int)(32u << 16);            // tensor_dim0[15:0]=32 (K cols)
  g1[2] = (int)((unsigned)rows << 16); // tensor_dim0[31:16]=0 | tensor_dim1[15:0]=rows
  g1[3] = (int)(32u << 16);            // tensor_dim1[31:16]=0 | tile_dim0=32
  g1[4] = 64;                          // tile_dim1=64, tile_dim2=0
  g1[5] = DDIM;                        // tensor_dim0_stride[31:0] = 768
  g1[6] = 0;                           // stride0[47:32]=0 | stride1[15:0]=0
  g1[7] = 0;
  i32x4 z4 = {};                       // group2/group3 NULL (<=2D tensor)
  i32x8 z8 = {};
  __builtin_amdgcn_tensor_load_to_lds(g0, g1, z4, z4, z8, 0);
}
#else
#define HAVE_TDM 0
#endif

// ----------------------------- WMMA GEMM: C[M,768] = A[M,768] @ W[768,768] --

__global__ __launch_bounds__(256) void gemm_xw_wmma(const bf16* __restrict__ A,
                                                    const bf16* __restrict__ W,
                                                    float* __restrict__ C, int M) {
  __shared__ bf16 lds_a[64][32];    // A tile, row-major (TDM destination)
  __shared__ bf16 lds_bT[128][32];  // W tile, transposed: [n][k]
  const int tid = threadIdx.x;
  const int lane = tid & 31, wave = tid >> 5;
  const int wm = wave >> 1, wn = wave & 1;        // 4 waves in M, 2 in N
  const int mblk = blockIdx.y * 64;
  const int nblk = blockIdx.x * 128;
  int rows = M - mblk;
  if (rows > 64) rows = 64;
  v8f acc[4];
  {
    v8f z = {};
#pragma unroll
    for (int j = 0; j < 4; ++j) acc[j] = z;
  }

  for (int k0 = 0; k0 < DDIM; k0 += 32) {
#if HAVE_TDM
    if (wave == 0) {  // one DMA per workgroup; TDM ignores EXEC
      tdm_load_tile_64x32_bf16(A + (size_t)mblk * DDIM + k0,
                               (unsigned)(uintptr_t)(void*)&lds_a[0][0], rows);
      __builtin_amdgcn_s_wait_tensorcnt(0);
    }
#else
    {  // stage 64x32 A tile manually
      int r = tid >> 2, ch = (tid & 3) * 8;
      int row = mblk + r;
      v8bf av = {};
      if (row < M) av = *(const v8bf*)(A + (size_t)row * DDIM + k0 + ch);
      *(v8bf*)&lds_a[r][ch] = av;
    }
#endif
    {  // stage 32x128 W tile, transposed into [n][k]
      int k = tid >> 3, nch = (tid & 7) * 16;
      const bf16* wp = W + (size_t)(k0 + k) * DDIM + nblk + nch;
      v8bf w0 = *(const v8bf*)wp;
      v8bf w1 = *(const v8bf*)(wp + 8);
#pragma unroll
      for (int i = 0; i < 8; ++i) {
        lds_bT[nch + i][k]     = w0[i];
        lds_bT[nch + 8 + i][k] = w1[i];
      }
    }
    __syncthreads();

    const int m = lane & 15, kh = lane >> 4;
    // A fragment (16x32): lanes 0-15 hold K[0:8]&K[16:24]; lanes 16-31 K[8:16]&K[24:32]
    v16bf afrag = cat8(*(const v8bf*)&lds_a[wm * 16 + m][kh * 8],
                       *(const v8bf*)&lds_a[wm * 16 + m][16 + kh * 8]);
#pragma unroll
    for (int j = 0; j < 4; ++j) {
      // B fragment (32x16): lanes 0-15 hold K[0:8]&K[16:24]; lanes 16-31 K[8:16]&K[24:32] of their column
      const bf16* bp = &lds_bT[wn * 64 + j * 16 + m][kh * 8];
      v16bf bfrag = cat8(*(const v8bf*)bp, *(const v8bf*)(bp + 16));
      acc[j] = __builtin_amdgcn_wmma_f32_16x16x32_bf16(false, afrag, false, bfrag,
                                                       (short)0, acc[j], false, false);
    }
    __syncthreads();
  }

  // C/D layout: vgpr r, lane l -> M = r + 8*(l>>4), N = l&15
  const int mrow = mblk + wm * 16 + (lane >> 4) * 8;
  const int ncol = nblk + wn * 64 + (lane & 15);
  if (mblk + 64 <= M) {  // uniform fast path: whole tile in range, no exec churn
#pragma unroll
    for (int j = 0; j < 4; ++j) {
#pragma unroll
      for (int r = 0; r < 8; ++r) {
        C[(size_t)(mrow + r) * DDIM + ncol + j * 16] = acc[j][r];
      }
    }
  } else {
#pragma unroll
    for (int j = 0; j < 4; ++j) {
#pragma unroll
      for (int r = 0; r < 8; ++r) {
        int row = mrow + r;
        if (row < M) C[(size_t)row * DDIM + ncol + j * 16] = acc[j][r];
      }
    }
  }
}

// ------------- WMMA GEMM: S[B,N] = Q[B,768] @ X[N,768]^T + sw*sparse -------

__global__ __launch_bounds__(128) void gemm_qx_wmma(const bf16* __restrict__ Q,
                                                    const bf16* __restrict__ X,
                                                    const float* __restrict__ sparse,
                                                    const float* __restrict__ sw,
                                                    float* __restrict__ S, int Nn) {
  const int lane = threadIdx.x & 31;
  const int wave = threadIdx.x >> 5;   // 4 waves == 64 query rows
  const int nblk = blockIdx.x * 64;
  const int m = wave * 16 + (lane & 15);
  const int kh = lane >> 4;
  v8f acc[4];
  {
    v8f z = {};
#pragma unroll
    for (int j = 0; j < 4; ++j) acc[j] = z;
  }
  for (int k0 = 0; k0 < DDIM; k0 += 32) {
    const bf16* qp = Q + (size_t)m * DDIM + k0;
    v16bf afrag = cat8(*(const v8bf*)(qp + kh * 8), *(const v8bf*)(qp + 16 + kh * 8));
#pragma unroll
    for (int j = 0; j < 4; ++j) {
      int n = nblk + j * 16 + (lane & 15);
      int nl = n < Nn ? n : Nn - 1;  // clamp loads; WMMA keeps full EXEC
      const bf16* xp = X + (size_t)nl * DDIM + k0 + kh * 8;
      v16bf bfrag = cat8(*(const v8bf*)xp, *(const v8bf*)(xp + 16));
      acc[j] = __builtin_amdgcn_wmma_f32_16x16x32_bf16(false, afrag, false, bfrag,
                                                       (short)0, acc[j], false, false);
    }
  }
  const float w = sw[0];
  const int rbase = wave * 16 + (lane >> 4) * 8;
  const int c16 = lane & 15;
#pragma unroll
  for (int j = 0; j < 4; ++j) {
    if (nblk + j * 16 < Nn) {  // subtile-uniform guard (Nn % 16 == 0)
      int n = nblk + j * 16 + c16;
#pragma unroll
      for (int r = 0; r < 8; ++r) {
        size_t o = (size_t)(rbase + r) * Nn + n;
        S[o] = acc[j][r] + w * sparse[o];
      }
    }
  }
}

// ----------------------------- top-k (one block per query row) -------------

__global__ __launch_bounds__(256) void topk_kernel(const float* __restrict__ S,
                                                   const int* __restrict__ qidx,
                                                   int* __restrict__ cand, int Nn) {
  __shared__ float svals[256];
  __shared__ int   sidx[256];
  __shared__ int   csel[TOPK];
  const int b = blockIdx.x;
  const int tid = threadIdx.x;
  const float* row = S + (size_t)b * Nn;
  for (int sel = 0; sel < TOPK; ++sel) {
    float best = -3.4e38f;
    int   bidx = 0x7fffffff;
    for (int n = tid; n < Nn; n += 256) {
      bool skip = false;
      for (int j = 0; j < sel; ++j)
        if (csel[j] == n) { skip = true; break; }
      if (skip) continue;
      float v = row[n];
      if (v > best || (v == best && n < bidx)) { best = v; bidx = n; }
    }
    svals[tid] = best; sidx[tid] = bidx;
    __syncthreads();
    for (int s = 128; s > 0; s >>= 1) {
      if (tid < s) {
        float v2 = svals[tid + s]; int i2 = sidx[tid + s];
        if (v2 > svals[tid] || (v2 == svals[tid] && i2 < sidx[tid])) {
          svals[tid] = v2; sidx[tid] = i2;
        }
      }
      __syncthreads();
    }
    if (tid == 0) csel[sel] = sidx[0];
    __syncthreads();
  }
  if (tid == 0) {
    int qi = qidx[b];
    bool has = false;
    for (int j = 0; j < TOPK; ++j)
      if (csel[j] == qi) has = true;
    if (!has) csel[TOPK - 1] = qi;
    for (int j = 0; j < TOPK; ++j) cand[b * TOPK + j] = csel[j];
  }
}

// ----------------------------- final scoring (1536 -> 1, ReLU) -------------

__global__ __launch_bounds__(256) void final_score_kernel(const float* __restrict__ G,
                                                          const float* __restrict__ Q,
                                                          const float* __restrict__ Wsc,
                                                          const float* __restrict__ bsc,
                                                          const int* __restrict__ cand,
                                                          float* __restrict__ outScore,
                                                          int* __restrict__ outIdx, int Bq) {
  int wid = threadIdx.x >> 5, lane = threadIdx.x & 31;
  int t = blockIdx.x * 8 + wid;
  if (t >= Bq * TOPK) return;
  int b = t / TOPK;
  int c = cand[t];
  const float* gr = G + (size_t)c * DDIM;
  const float* qr = Q + (size_t)b * DDIM;
  float s = 0.f;
  for (int j = lane; j < 2 * DDIM; j += 32) {
    float f = (j < DDIM) ? gr[j] : qr[j - DDIM];
    s += f * Wsc[j];
  }
#pragma unroll
  for (int off = 16; off > 0; off >>= 1) s += __shfl_xor(s, off, 32);
  if (lane == 0) {
    float r = s + bsc[0];
    outScore[t] = r > 0.f ? r : 0.f;
    outIdx[t] = c;
  }
}

// ----------------------------- host launcher -------------------------------

extern "C" void kernel_launch(void* const* d_in, const int* in_sizes, int n_in,
                              void* d_out, int out_size, void* d_ws, size_t ws_size,
                              hipStream_t stream) {
  const float* q      = (const float*)d_in[0];
  const float* xnames = (const float*)d_in[1];
  const float* sparse = (const float*)d_in[2];
  const int*   qidx   = (const int*)d_in[3];
  const int*   eidx   = (const int*)d_in[4];
  const float* sw     = (const float*)d_in[5];
  const float* W1     = (const float*)d_in[6];
  const float* b1     = (const float*)d_in[7];
  const float* W2     = (const float*)d_in[8];
  const float* b2     = (const float*)d_in[9];
  const float* Wsc    = (const float*)d_in[10];
  const float* bsc    = (const float*)d_in[11];

  const int Bq = in_sizes[0] / DDIM;   // 64
  const int Nn = in_sizes[1] / DDIM;   // 50000
  const int E  = in_sizes[4] / 2;      // 500000
  const int* esrc = eidx;
  const int* edst = eidx + E;

  char* ws = (char*)d_ws;
  size_t off = 0;
  auto alloc = [&](size_t bytes) {
    size_t p = off;
    off += (bytes + 255) & ~(size_t)255;
    return p;
  };
  float* deg  = (float*)(ws + alloc((size_t)Nn * 4));
  float* dinv = (float*)(ws + alloc((size_t)Nn * 4));
  bf16*  xbf  = (bf16*)(ws + alloc((size_t)Nn * DDIM * 2));  // names bf16, later reused as layer-2 input
  bf16*  wbf  = (bf16*)(ws + alloc((size_t)DDIM * DDIM * 2));
  bf16*  qbf  = (bf16*)(ws + alloc((size_t)Bq * DDIM * 2));
  float* H    = (float*)(ws + alloc((size_t)Nn * DDIM * 4));
  float* G    = (float*)(ws + alloc((size_t)Nn * DDIM * 4));
  float* Sbuf = (float*)(ws + alloc((size_t)Bq * Nn * 4));
  int*   cand = (int*)(ws + alloc((size_t)Bq * TOPK * 4));

  const long nx = (long)Nn * DDIM;

  // degrees + symmetric norm
  deg_init_kernel<<<(Nn + 255) / 256, 256, 0, stream>>>(deg, Nn);
  deg_edge_kernel<<<(E + 255) / 256, 256, 0, stream>>>(edst, deg, E);
  dinv_kernel<<<(Nn + 255) / 256, 256, 0, stream>>>(deg, dinv, Nn);

  // bf16 staging
  cvt_bf16_kernel<<<4096, 256, 0, stream>>>(xnames, xbf, nx, 0);
  cvt_bf16_kernel<<<64, 256, 0, stream>>>(q, qbf, (long)Bq * DDIM, 0);

  // retrieval: bert score (WMMA) + fused sparse score, then top-k w/ gold forcing
  gemm_qx_wmma<<<(Nn + 63) / 64, 128, 0, stream>>>(qbf, xbf, sparse, sw, Sbuf, Nn);
  topk_kernel<<<Bq, 256, 0, stream>>>(Sbuf, qidx, cand, Nn);

  // GCN layer 1
  cvt_bf16_kernel<<<512, 256, 0, stream>>>(W1, wbf, (long)DDIM * DDIM, 0);
  dim3 g1((DDIM + 127) / 128, (Nn + 63) / 64);
  gemm_xw_wmma<<<g1, 256, 0, stream>>>(xbf, wbf, H, Nn);
  ginit_kernel<<<4096, 256, 0, stream>>>(H, dinv, b1, G, nx);
  scatter_kernel<<<(E + 7) / 8, 256, 0, stream>>>(H, G, esrc, edst, dinv, E);

  // ReLU + bf16 (reuse xbf buffer: retrieval is done with it)
  cvt_bf16_kernel<<<4096, 256, 0, stream>>>(G, xbf, nx, 1);

  // GCN layer 2
  cvt_bf16_kernel<<<512, 256, 0, stream>>>(W2, wbf, (long)DDIM * DDIM, 0);
  gemm_xw_wmma<<<g1, 256, 0, stream>>>(xbf, wbf, H, Nn);
  ginit_kernel<<<4096, 256, 0, stream>>>(H, dinv, b2, G, nx);
  scatter_kernel<<<(E + 7) / 8, 256, 0, stream>>>(H, G, esrc, edst, dinv, E);

  // final candidate scoring -> d_out = [scores (B*K f32) | indices (B*K i32)]
  float* outScore = (float*)d_out;
  int*   outIdx   = (int*)d_out + (size_t)Bq * TOPK;
  final_score_kernel<<<(Bq * TOPK + 7) / 8, 256, 0, stream>>>(G, q, Wsc, bsc, cand,
                                                              outScore, outIdx, Bq);
}
